// TrajectoryDecoder_8650064134853
// MI455X (gfx1250) — matrix operand order repaired and checked by reading, not compiled, not run on hardware
//
#include <hip/hip_runtime.h>
#include <hip/hip_bf16.h>

#define H 256
#define H3 768
#define T_FUT 30
#define KM 6
#define BKROWS 12288
#define ROWS_PER_WG 64
#define NWG (BKROWS / ROWS_PER_WG)   // 192
#define THREADS 128                  // 4 wave32 per WG, 16 rows per wave

typedef __attribute__((ext_vector_type(16))) __bf16 v16bf;
typedef __attribute__((ext_vector_type(8)))  __bf16 v8bf;
typedef __attribute__((ext_vector_type(8)))  float  v8f;
typedef __attribute__((ext_vector_type(4)))  float  v4f;

// bf16 workspace layout (element offsets)
#define OFF_WHH0  0u
#define OFF_WIH1  196608u
#define OFF_WHH1  393216u
#define OFF_WINIT 589824u           // 256 rows x stride 264
#define WINIT_STRIDE 264

static __device__ __forceinline__ v8f wmma_bf16(v16bf a, v16bf b, v8f c) {
  // D = A(16x32) * B(32x16) + C, f32 accumulate
  return __builtin_amdgcn_wmma_f32_16x16x32_bf16(false, a, false, b, (short)0, c, false, false);
}

// ---- branch-free hardware transcendentals (GRU inputs are small-magnitude;
// ---- exp2 underflow/overflow saturates exactly as the math requires) ----
#define LOG2E 1.4426950408889634f

static __device__ __forceinline__ float sigmoidf_(float x) {
  return __builtin_amdgcn_rcpf(1.0f + __builtin_amdgcn_exp2f(-LOG2E * x));
}

#if __has_builtin(__builtin_amdgcn_tanhf)
static __device__ __forceinline__ float tanhf_(float x) {
  return __builtin_amdgcn_tanhf(x);         // v_tanh_f32 (CDNA5 TRANS op)
}
#else
static __device__ __forceinline__ float tanhf_(float x) {
  // tanh(x) = 2*sigmoid(2x) - 1
  return 2.0f * __builtin_amdgcn_rcpf(1.0f + __builtin_amdgcn_exp2f(-2.0f * LOG2E * x)) - 1.0f;
}
#endif

// A fragment (16x32 bf16) from a row-major f32 source (stride in floats).
// ISA 16-bit A layout: lanes 0-15 -> M=lane, K = k0+{0..7} and k0+16+{0..7};
// lanes 16-31 -> same M, K offset +8.
static __device__ __forceinline__ v16bf a_frag_f32(const float* base, int stride,
                                                   int k0, int lane) {
  const int m    = lane & 15;
  const int hoff = (lane >> 4) << 3;        // 0 or 8
  const float* p = base + m * stride + k0 + hoff;
  v4f a0 = *(const v4f*)(p + 0);
  v4f a1 = *(const v4f*)(p + 4);
  v4f a2 = *(const v4f*)(p + 16);
  v4f a3 = *(const v4f*)(p + 20);
  v16bf r;
#pragma unroll
  for (int i = 0; i < 4; ++i) {
    r[i]      = (__bf16)a0[i];
    r[4 + i]  = (__bf16)a1[i];
    r[8 + i]  = (__bf16)a2[i];
    r[12 + i] = (__bf16)a3[i];
  }
  return r;
}

// A fragment for h_init: rows map bk -> context[bk/KM]
static __device__ __forceinline__ v16bf a_frag_ctx(const float* ctx, int gr0,
                                                   int k0, int lane) {
  const int m    = lane & 15;
  const int hoff = (lane >> 4) << 3;
  const float* p = ctx + (size_t)((gr0 + m) / KM) * H + k0 + hoff;
  v4f a0 = *(const v4f*)(p + 0);
  v4f a1 = *(const v4f*)(p + 4);
  v4f a2 = *(const v4f*)(p + 16);
  v4f a3 = *(const v4f*)(p + 20);
  v16bf r;
#pragma unroll
  for (int i = 0; i < 4; ++i) {
    r[i]      = (__bf16)a0[i];
    r[4 + i]  = (__bf16)a1[i];
    r[8 + i]  = (__bf16)a2[i];
    r[12 + i] = (__bf16)a3[i];
  }
  return r;
}

// B fragment (32x16 bf16) from row-major bf16 weights W[n][k] (gh = h @ W^T).
// ISA B layout: lane = N column (lane&15);
// lanes 0-15 hold K = k0+{0..7} and k0+16+{0..7}, lanes 16-31 same with K offset +8.
static __device__ __forceinline__ v16bf b_frag(const __bf16* W, int row0, int stride,
                                               int k0, int lane) {
  const int n    = lane & 15;
  const int koff = k0 + ((lane >> 4) << 3);  // 0 or 8
  const __bf16* p = W + (size_t)(row0 + n) * stride + koff;
  v8bf lo = *(const v8bf*)(p);
  v8bf hi = *(const v8bf*)(p + 16);
  return __builtin_shufflevector(lo, hi, 0,1,2,3,4,5,6,7,8,9,10,11,12,13,14,15);
}

// ---------------- weight conversion kernels ----------------
__global__ void cvt_f32_bf16(const float* __restrict__ src, __bf16* __restrict__ dst, int n) {
  int i = blockIdx.x * blockDim.x + threadIdx.x;
  if (i < n) dst[i] = (__bf16)src[i];
}

__global__ void cvt_winit(const float* __restrict__ src, __bf16* __restrict__ dst) {
  int i = blockIdx.x * blockDim.x + threadIdx.x;
  if (i < H * 258) {
    int r = i / 258, c = i % 258;
    dst[r * WINIT_STRIDE + c] = (__bf16)src[i];
  }
}

// ---------------- main persistent GRU decoder ----------------
__global__ __launch_bounds__(THREADS)
void traj_gru_kernel(const float* __restrict__ context,
                     const float* __restrict__ goals,
                     const float* __restrict__ hinit_w,   // f32 (256,258)
                     const float* __restrict__ hinit_b,
                     const float* __restrict__ w_ih0,     // (768,2)
                     const float* __restrict__ b_ih0,
                     const float* __restrict__ b_hh0,
                     const float* __restrict__ b_ih1,
                     const float* __restrict__ b_hh1,
                     const float* __restrict__ out_w,     // (2,256)
                     const float* __restrict__ out_b,     // (2,)
                     const __bf16* __restrict__ Whh0,
                     const __bf16* __restrict__ Wih1,
                     const __bf16* __restrict__ Whh1,
                     const __bf16* __restrict__ Winit,
                     float* __restrict__ out)
{
  extern __shared__ char smem_raw[];
  float* sh0  = (float*)smem_raw;                    // 64*256 f32
  float* sh1  = sh0 + ROWS_PER_WG * H;               // 64*256 f32
  float* sxv  = sh1 + ROWS_PER_WG * H;               // 64*2 (delta feedback)
  float* spos = sxv + ROWS_PER_WG * 2;               // 64*2

  const int lane    = threadIdx.x & 31;
  const int wave    = threadIdx.x >> 5;
  const int r0w     = wave * 16;                     // wave's row base in WG
  const int gr0     = blockIdx.x * ROWS_PER_WG + r0w;// global bk row base
  const int nloc    = lane & 15;
  const int halfsel = lane >> 4;

  float* h0w = sh0 + r0w * H;   // this wave's private 16xH state tiles
  float* h1w = sh1 + r0w * H;

  // ---------------- h_init = tanh([ctx,g] @ Winit^T + b) ----------------
  for (int nc = 0; nc < 16; ++nc) {
    const int nb = nc * 16;
    v8f acc = {};
#pragma unroll
    for (int kt = 0; kt < 8; ++kt) {
      v16bf a = a_frag_ctx(context, gr0, kt * 32, lane);
      v16bf b = b_frag(Winit, nb, WINIT_STRIDE, kt * 32, lane);
      acc = wmma_bf16(a, b, acc);
    }
    const int col = nb + nloc;
    const float wg0 = hinit_w[col * 258 + 256];
    const float wg1 = hinit_w[col * 258 + 257];
    const float bb  = hinit_b[col];
#pragma unroll
    for (int v = 0; v < 8; ++v) {
      const int rloc = v + halfsel * 8;
      const int bk   = gr0 + rloc;
      const float g0 = goals[bk * 2 + 0];
      const float g1 = goals[bk * 2 + 1];
      const float hv = tanhf_(acc[v] + g0 * wg0 + g1 * wg1 + bb);
      h0w[rloc * H + col] = hv;
      h1w[rloc * H + col] = hv;
    }
  }
  {
    const int rloc = r0w + (lane & 15);
    sxv[rloc * 2 + halfsel]  = 0.0f;
    spos[rloc * 2 + halfsel] = 0.0f;
  }

  // ---------------- 30 recurrent steps ----------------
  for (int t = 0; t < T_FUT; ++t) {
    // ===== layer 0: gh0 = h0 @ Whh0^T (r,z,n), gi0 from 2-wide x =====
    v16bf A0[8];
#pragma unroll
    for (int kt = 0; kt < 8; ++kt) A0[kt] = a_frag_f32(h0w, H, kt * 32, lane);

    for (int nc = 0; nc < 16; ++nc) {
      const int nb = nc * 16;
      v8f ar = {}, az = {}, an = {};
#pragma unroll
      for (int kt = 0; kt < 8; ++kt) {
        v16bf br = b_frag(Whh0, nb,       H, kt * 32, lane);
        v16bf bz = b_frag(Whh0, H  + nb,  H, kt * 32, lane);
        v16bf bn = b_frag(Whh0, 2*H + nb, H, kt * 32, lane);
        ar = wmma_bf16(A0[kt], br, ar);
        az = wmma_bf16(A0[kt], bz, az);
        an = wmma_bf16(A0[kt], bn, an);
      }
      const int col = nb + nloc;
      const float bir = b_ih0[col], biz = b_ih0[H + col], bin_ = b_ih0[2*H + col];
      const float bhr = b_hh0[col], bhz = b_hh0[H + col], bhn  = b_hh0[2*H + col];
      const float wr0 = w_ih0[col * 2],           wr1 = w_ih0[col * 2 + 1];
      const float wz0 = w_ih0[(H + col) * 2],     wz1 = w_ih0[(H + col) * 2 + 1];
      const float wn0 = w_ih0[(2*H + col) * 2],   wn1 = w_ih0[(2*H + col) * 2 + 1];
#pragma unroll
      for (int v = 0; v < 8; ++v) {
        const int rloc = v + halfsel * 8;
        const float x0 = sxv[(r0w + rloc) * 2 + 0];
        const float x1 = sxv[(r0w + rloc) * 2 + 1];
        const float rr = sigmoidf_(ar[v] + bhr + x0 * wr0 + x1 * wr1 + bir);
        const float zz = sigmoidf_(az[v] + bhz + x0 * wz0 + x1 * wz1 + biz);
        const float nn = tanhf_(x0 * wn0 + x1 * wn1 + bin_ + rr * (an[v] + bhn));
        const float ho = h0w[rloc * H + col];
        h0w[rloc * H + col] = (1.0f - zz) * nn + zz * ho;
      }
    }

    // ===== layer 1: gi1 = h0_new @ Wih1^T, gh1 = h1 @ Whh1^T =====
    v16bf A0n[8], A1[8];
#pragma unroll
    for (int kt = 0; kt < 8; ++kt) {
      A0n[kt] = a_frag_f32(h0w, H, kt * 32, lane);
      A1[kt]  = a_frag_f32(h1w, H, kt * 32, lane);
    }

    for (int nc = 0; nc < 16; ++nc) {
      const int nb = nc * 16;
      v8f air = {}, aiz = {}, ain = {}, ahr = {}, ahz = {}, ahn = {};
#pragma unroll
      for (int kt = 0; kt < 8; ++kt) {
        v16bf bir_ = b_frag(Wih1, nb,       H, kt * 32, lane);
        v16bf biz_ = b_frag(Wih1, H  + nb,  H, kt * 32, lane);
        v16bf bin2 = b_frag(Wih1, 2*H + nb, H, kt * 32, lane);
        v16bf bhr_ = b_frag(Whh1, nb,       H, kt * 32, lane);
        v16bf bhz_ = b_frag(Whh1, H  + nb,  H, kt * 32, lane);
        v16bf bhn_ = b_frag(Whh1, 2*H + nb, H, kt * 32, lane);
        air = wmma_bf16(A0n[kt], bir_, air);
        aiz = wmma_bf16(A0n[kt], biz_, aiz);
        ain = wmma_bf16(A0n[kt], bin2, ain);
        ahr = wmma_bf16(A1[kt],  bhr_, ahr);
        ahz = wmma_bf16(A1[kt],  bhz_, ahz);
        ahn = wmma_bf16(A1[kt],  bhn_, ahn);
      }
      const int col = nb + nloc;
      const float bi1r = b_ih1[col], bi1z = b_ih1[H + col], bi1n = b_ih1[2*H + col];
      const float bh1r = b_hh1[col], bh1z = b_hh1[H + col], bh1n = b_hh1[2*H + col];
#pragma unroll
      for (int v = 0; v < 8; ++v) {
        const int rloc = v + halfsel * 8;
        const float rr = sigmoidf_(air[v] + bi1r + ahr[v] + bh1r);
        const float zz = sigmoidf_(aiz[v] + bi1z + ahz[v] + bh1z);
        const float nn = tanhf_(ain[v] + bi1n + rr * (ahn[v] + bh1n));
        const float ho = h1w[rloc * H + col];
        h1w[rloc * H + col] = (1.0f - zz) * nn + zz * ho;
      }
    }

    // ===== delta = h1 @ out_w^T + out_b ; pos += delta ; write raw traj =====
    {
      const int rloc = lane & 15;      // one row per lane pair
      const int c    = halfsel;        // x/y component
      const float* hrow = h1w + rloc * H;
      const float* wrow = out_w + c * H;
      float s = out_b[c];
#pragma unroll 8
      for (int j = 0; j < H; j += 4) {
        v4f hv = *(const v4f*)(hrow + j);
        v4f wv = *(const v4f*)(wrow + j);
        s += hv[0]*wv[0] + hv[1]*wv[1] + hv[2]*wv[2] + hv[3]*wv[3];
      }
      const int lrow = r0w + rloc;
      sxv[lrow * 2 + c] = s;                       // feedback x for next step
      const float p = spos[lrow * 2 + c] + s;
      spos[lrow * 2 + c] = p;
      const int bk = gr0 + rloc;
      out[(size_t)bk * (T_FUT * 2) + t * 2 + c] = p;
    }
  }
}

// ---------------- ramp correction: traj += ramp * (g - traj_last) ----------------
__global__ void ramp_kernel(const float* __restrict__ goals, float* __restrict__ out) {
  int i = blockIdx.x * blockDim.x + threadIdx.x;
  if (i >= BKROWS * 2) return;
  const int bk = i >> 1, c = i & 1;
  float* p = out + (size_t)bk * (T_FUT * 2) + c;
  const float last = p[(T_FUT - 1) * 2];
  const float err  = goals[bk * 2 + c] - last;
#pragma unroll
  for (int t = 0; t < T_FUT; ++t) {
    p[t * 2] += ((float)(t + 1) / (float)T_FUT) * err;
  }
}

extern "C" void kernel_launch(void* const* d_in, const int* in_sizes, int n_in,
                              void* d_out, int out_size, void* d_ws, size_t ws_size,
                              hipStream_t stream) {
  const float* context = (const float*)d_in[0];
  const float* goals   = (const float*)d_in[1];
  const float* hinit_w = (const float*)d_in[2];
  const float* hinit_b = (const float*)d_in[3];
  const float* w_ih0   = (const float*)d_in[4];
  const float* w_hh0   = (const float*)d_in[5];
  const float* b_ih0   = (const float*)d_in[6];
  const float* b_hh0   = (const float*)d_in[7];
  const float* w_ih1   = (const float*)d_in[8];
  const float* w_hh1   = (const float*)d_in[9];
  const float* b_ih1   = (const float*)d_in[10];
  const float* b_hh1   = (const float*)d_in[11];
  const float* out_w   = (const float*)d_in[12];
  const float* out_b   = (const float*)d_in[13];
  float* out = (float*)d_out;

  __bf16* ws   = (__bf16*)d_ws;
  __bf16* Whh0 = ws + OFF_WHH0;
  __bf16* Wih1 = ws + OFF_WIH1;
  __bf16* Whh1 = ws + OFF_WHH1;
  __bf16* Winit= ws + OFF_WINIT;

  const int nW = H3 * H;  // 196608
  cvt_f32_bf16<<<(nW + 255) / 256, 256, 0, stream>>>(w_hh0, Whh0, nW);
  cvt_f32_bf16<<<(nW + 255) / 256, 256, 0, stream>>>(w_ih1, Wih1, nW);
  cvt_f32_bf16<<<(nW + 255) / 256, 256, 0, stream>>>(w_hh1, Whh1, nW);
  cvt_winit<<<(H * 258 + 255) / 256, 256, 0, stream>>>(hinit_w, Winit);

  const size_t smem = (size_t)(ROWS_PER_WG * H * 2 + ROWS_PER_WG * 2 * 2) * sizeof(float); // 132096 B
  traj_gru_kernel<<<NWG, THREADS, smem, stream>>>(
      context, goals, hinit_w, hinit_b, w_ih0, b_ih0, b_hh0, b_ih1, b_hh1,
      out_w, out_b, Whh0, Wih1, Whh1, Winit, out);

  ramp_kernel<<<(BKROWS * 2 + 255) / 256, 256, 0, stream>>>(goals, out);
}
